// _TransformerBlock_21260088115361
// MI455X (gfx1250) — matrix operand order repaired and checked by reading, not compiled, not run on hardware
//
#include <hip/hip_runtime.h>
#include <math.h>

// ---------------------------------------------------------------------------
// Transformer block for MI455X (gfx1250, wave32, WMMA bf16 16x16x32).
// B=2, L=2048, D=1024, H=16, DH=64, FF=4096.  All GEMMs run through
// v_wmma_f32_16x16x32_bf16; attention is flash-style (no [B,H,L,L] buffer).
// Tile staging uses GLOBAL_LOAD_ASYNC_TO_LDS_B128 (ASYNCcnt) when the
// toolchain exposes the builtin, with a plain load+ds_store fallback.
// Workspace usage: ~120 MB carved from d_ws.
// ---------------------------------------------------------------------------

typedef unsigned short u16;
typedef unsigned int   v4u   __attribute__((ext_vector_type(4)));
typedef __bf16         v16bf __attribute__((ext_vector_type(16)));
typedef float          v8f   __attribute__((ext_vector_type(8)));
typedef int            i4    __attribute__((vector_size(16)));   // matches builtin V4i

union FragU { v16bf v; v4u q[2]; };

#if defined(__has_builtin)
#if __has_builtin(__builtin_amdgcn_global_load_async_to_lds_b128)
#define HAVE_ASYNC_LDS 1
#endif
#endif
#ifndef HAVE_ASYNC_LDS
#define HAVE_ASYNC_LDS 0
#endif

// 16-byte global -> LDS copy (per active lane).
__device__ __forceinline__ void cp128(const u16* g, u16* l) {
#if HAVE_ASYNC_LDS
    __builtin_amdgcn_global_load_async_to_lds_b128(
        (__attribute__((address_space(1))) i4*)(__attribute__((address_space(1))) void*)(void*)g,
        (__attribute__((address_space(3))) i4*)(__attribute__((address_space(3))) void*)(void*)l,
        /*offset=*/0, /*cpol=*/0);
#else
    *(v4u*)l = *(const v4u*)g;
#endif
}

__device__ __forceinline__ void async_wait() {
#if HAVE_ASYNC_LDS
#if __has_builtin(__builtin_amdgcn_s_wait_asynccnt)
    __builtin_amdgcn_s_wait_asynccnt(0);
#else
    asm volatile("s_wait_asynccnt 0x0" ::: "memory");
#endif
#endif
}

__device__ __forceinline__ v8f zero8() {
    v8f z = {0.f, 0.f, 0.f, 0.f, 0.f, 0.f, 0.f, 0.f};
    return z;
}

__device__ __forceinline__ u16 f2bf(float f) {
    unsigned int u = __float_as_uint(f);
    unsigned int r = u + 0x7FFFu + ((u >> 16) & 1u);   // round-to-nearest-even
    return (u16)(r >> 16);
}

__device__ __forceinline__ v8f wmma_bf16(v16bf a, v16bf b, v8f c) {
    return __builtin_amdgcn_wmma_f32_16x16x32_bf16(
        /*neg_a=*/false, a, /*neg_b=*/false, b,
        /*c_mod=*/(short)0, c, /*reuse_a=*/false, /*reuse_b=*/false);
}

// A-fragment (16x32 bf16, M = lane%16): lanes 0-15 hold K={0..7,16..23},
// lanes 16-31 hold K={8..15,24..31}  -> two ds_load_b128.
__device__ __forceinline__ v16bf frag_a(const u16* lds, int row, int ld, int k0, int lane) {
    int half = (lane >> 4) & 1;
    const u16* p = lds + row * ld + k0 + 8 * half;
    FragU f;
    f.q[0] = *(const v4u*)(p);
    f.q[1] = *(const v4u*)(p + 16);
    return f.v;
}

// B-fragment (32x16 bf16, stored transposed [N][K] in LDS; N = lane%16):
// lanes 0-15 hold K={0..7,16..23}, lanes 16-31 hold K={8..15,24..31} -> two ds_load_b128.
__device__ __forceinline__ v16bf frag_b(const u16* lds, int row, int ld, int k0, int lane) {
    int half = (lane >> 4) & 1;
    const u16* p = lds + row * ld + k0 + 8 * half;
    FragU f;
    f.q[0] = *(const v4u*)(p);
    f.q[1] = *(const v4u*)(p + 16);
    return f.v;
}

// ---------------------------------------------------------------------------
// Weight prep: f32 [K][N] -> bf16 [N][K] (transpose + convert).
// ---------------------------------------------------------------------------
__global__ __launch_bounds__(256) void cvt_transpose_kernel(
    const float* __restrict__ in, u16* __restrict__ out, int K, int N) {
    int idx = blockIdx.x * 256 + threadIdx.x;
    if (idx >= K * N) return;
    int k = idx / N, n = idx % N;
    out[(size_t)n * K + k] = f2bf(in[idx]);
}

// ---------------------------------------------------------------------------
// LayerNorm over D=1024, optional +pos_embed, bf16 output.  1 block per row.
// ---------------------------------------------------------------------------
__global__ __launch_bounds__(256) void ln_kernel(
    const float* __restrict__ x, const float* __restrict__ g,
    const float* __restrict__ bt, const float* __restrict__ pos,
    u16* __restrict__ out, int addpos) {
    const int row = blockIdx.x, t = threadIdx.x;
    const int lane = t & 31, w = t >> 5;
    const float* xr = x + (size_t)row * 1024;
    float4 v = *(const float4*)(xr + t * 4);
    float vv[4] = {v.x, v.y, v.z, v.w};
    float s  = vv[0] + vv[1] + vv[2] + vv[3];
    float s2 = vv[0]*vv[0] + vv[1]*vv[1] + vv[2]*vv[2] + vv[3]*vv[3];
#pragma unroll
    for (int off = 16; off > 0; off >>= 1) {
        s  += __shfl_xor(s,  off, 32);
        s2 += __shfl_xor(s2, off, 32);
    }
    __shared__ float sm[8], sq[8];
    if (lane == 0) { sm[w] = s; sq[w] = s2; }
    __syncthreads();
    float ts = 0.f, ts2 = 0.f;
#pragma unroll
    for (int i = 0; i < 8; ++i) { ts += sm[i]; ts2 += sq[i]; }
    const float mean = ts * (1.f / 1024.f);
    const float var  = ts2 * (1.f / 1024.f) - mean * mean;
    const float rinv = rsqrtf(var + 1e-5f);
    const float* pr = addpos ? (pos + (size_t)(row & 2047) * 1024) : nullptr;
#pragma unroll
    for (int j = 0; j < 4; ++j) {
        int i = t * 4 + j;
        float val = (vv[j] - mean) * rinv * g[i] + bt[i];
        if (addpos) val += pr[i];
        out[(size_t)row * 1024 + i] = f2bf(val);
    }
}

// ---------------------------------------------------------------------------
// Tiled WMMA GEMM: C[M,N] = A[M,K](bf16,row-major) x Bt[N,K](bf16)^T.
// Block tile 128x128; 8 waves as 4x2 grid, each wave owns a 32x64 sub-tile
// (2 A-frags x 4 B-frags -> 8 WMMA from 6 fragment loads).  K staged 32 at
// a time through double-buffered LDS filled by async global->LDS copies
// issued before the current chunk's WMMA work (DMA overlaps math).
// MODE 0: bf16 scatter -> [B,H,L,DH]      (Q, K projections)
// MODE 1: bf16 scatter -> [B,H,DH,L]      (V projection, transposed)
// MODE 2: bf16 row-major, +bias, exact GELU (W1)
// MODE 3: f32  row-major, +bias, +residual  (Wo, W2/final)
// ---------------------------------------------------------------------------
template <int MODE>
__global__ __launch_bounds__(256) void gemm_bf16_kernel(
    const u16* __restrict__ A, const u16* __restrict__ Bt,
    void* __restrict__ outp, const float* __restrict__ bias,
    const float* __restrict__ res, int N, int K) {
    __shared__ __align__(16) u16 As[2][128 * 40];
    __shared__ __align__(16) u16 Bs[2][128 * 40];
    const int t = threadIdx.x, lane = t & 31, w = t >> 5;
    const int half = lane >> 4, nn = lane & 15;
    const int m0 = blockIdx.y * 128, n0 = blockIdx.x * 128;
    const int wr = (w >> 1) * 32;          // wave row base within block tile
    const int wc = (w & 1) * 64;           // wave col base within block tile
    const int lrow = t >> 1, loff = (t & 1) * 16;   // 16 bf16 per thread per tile

    v8f acc[8];
#pragma unroll
    for (int i = 0; i < 8; ++i) acc[i] = zero8();

    const u16* gA = A  + (size_t)(m0 + lrow) * K + loff;
    const u16* gB = Bt + (size_t)(n0 + lrow) * K + loff;

    // prologue: fill buffer 0 with chunk 0
    cp128(gA,     As[0] + lrow * 40 + loff);
    cp128(gA + 8, As[0] + lrow * 40 + loff + 8);
    cp128(gB,     Bs[0] + lrow * 40 + loff);
    cp128(gB + 8, Bs[0] + lrow * 40 + loff + 8);
    async_wait();
    __syncthreads();

    int cur = 0;
    for (int kc = 0; kc < K; kc += 32) {
        if (kc + 32 < K) {                 // kick off next chunk's DMA first
            int nxt = cur ^ 1;
            cp128(gA + kc + 32,     As[nxt] + lrow * 40 + loff);
            cp128(gA + kc + 32 + 8, As[nxt] + lrow * 40 + loff + 8);
            cp128(gB + kc + 32,     Bs[nxt] + lrow * 40 + loff);
            cp128(gB + kc + 32 + 8, Bs[nxt] + lrow * 40 + loff + 8);
        }
        v16bf a0 = frag_a(As[cur], wr + nn,      40, 0, lane);
        v16bf a1 = frag_a(As[cur], wr + 16 + nn, 40, 0, lane);
#pragma unroll
        for (int bn = 0; bn < 4; ++bn) {
            v16bf bf_ = frag_b(Bs[cur], wc + bn * 16 + nn, 40, 0, lane);
            acc[bn]     = wmma_bf16(a0, bf_, acc[bn]);
            acc[4 + bn] = wmma_bf16(a1, bf_, acc[4 + bn]);
        }
        async_wait();
        __syncthreads();
        cur ^= 1;
    }

#pragma unroll
    for (int ar = 0; ar < 2; ++ar) {
#pragma unroll
        for (int bn = 0; bn < 4; ++bn) {
#pragma unroll
            for (int r = 0; r < 8; ++r) {
                // C layout: VGPR r -> row r + 8*(lane/16); lane%16 -> column
                int m = m0 + wr + ar * 16 + r + 8 * half;
                int n = n0 + wc + bn * 16 + nn;
                float v = acc[ar * 4 + bn][r];
                if (MODE == 0) {                   // -> [B,H,L,DH]
                    int b = m >> 11, l = m & 2047, h = n >> 6, dh = n & 63;
                    ((u16*)outp)[(((size_t)(b * 16 + h)) * 2048 + l) * 64 + dh] = f2bf(v);
                } else if (MODE == 1) {            // -> [B,H,DH,L]
                    int b = m >> 11, l = m & 2047, h = n >> 6, dh = n & 63;
                    ((u16*)outp)[(((size_t)(b * 16 + h)) * 64 + dh) * 2048 + l] = f2bf(v);
                } else if (MODE == 2) {            // bias + exact GELU -> bf16
                    float z = v + bias[n];
                    z = 0.5f * z * (1.0f + erff(z * 0.70710678118654752f));
                    ((u16*)outp)[(size_t)m * N + n] = f2bf(z);
                } else {                           // bias + residual -> f32
                    ((float*)outp)[(size_t)m * N + n] = v + bias[n] + res[(size_t)m * N + n];
                }
            }
        }
    }
}

// ---------------------------------------------------------------------------
// Flash attention: grid = (16 q-blocks, 32 b*h).  Block = 256 thr = 8 waves.
// Each wave owns 16 query rows; 64-key blocks streamed through LDS via the
// async global->LDS path.  Q:[BH,L,DH] K:[BH,L,DH] Vt:[BH,DH,L] bf16.
// ---------------------------------------------------------------------------
__global__ __launch_bounds__(256) void attn_kernel(
    const u16* __restrict__ Q, const u16* __restrict__ Kg,
    const u16* __restrict__ Vt, u16* __restrict__ out) {
    __shared__ __align__(16) u16 Qs[128 * 72];
    __shared__ __align__(16) u16 Ks[64 * 72];
    __shared__ __align__(16) u16 Vts[64 * 72];
    __shared__ __align__(16) u16 Ps[128 * 72];

    const int t = threadIdx.x, lane = t & 31, w = t >> 5;
    const int half = lane >> 4, nn = lane & 15;
    const int qblk = blockIdx.x, bh = blockIdx.y;
    const int b = bh >> 4, h = bh & 15;
    const size_t base = (size_t)bh * 2048 * 64;   // per-(b,h) base for Q/K/Vt

    {   // stage the 128x64 Q tile once (async DMA into LDS)
        int row = t >> 1, col = (t & 1) * 32;
        const u16* g = Q + base + (size_t)(qblk * 128 + row) * 64 + col;
        u16* s = Qs + row * 72 + col;
        cp128(g, s); cp128(g + 8, s + 8); cp128(g + 16, s + 16); cp128(g + 24, s + 24);
    }

    float m_run[8], l_run[8];
#pragma unroll
    for (int r = 0; r < 8; ++r) { m_run[r] = -1e30f; l_run[r] = 0.f; }
    v8f o_acc[4];
#pragma unroll
    for (int i = 0; i < 4; ++i) o_acc[i] = zero8();

    const int krow = t >> 2, kcol = (t & 3) * 16;
    for (int kb = 0; kb < 2048; kb += 64) {
        {   // stage 64x64 K block and 64(dh) x 64(keys) Vt block
            const u16* gk = Kg + base + (size_t)(kb + krow) * 64 + kcol;
            const u16* gv = Vt + base + (size_t)krow * 2048 + kb + kcol;
            u16* sk = Ks  + krow * 72 + kcol;
            u16* sv = Vts + krow * 72 + kcol;
            cp128(gk, sk); cp128(gk + 8, sk + 8);
            cp128(gv, sv); cp128(gv + 8, sv + 8);
        }
        async_wait();
        __syncthreads();

        // S = Q(16x64) . K^T  -> four 16x16 score tiles, K-dim = 64 (2 steps)
        v16bf a0 = frag_a(Qs, w * 16 + nn, 72, 0, lane);
        v16bf a1 = frag_a(Qs, w * 16 + nn, 72, 32, lane);
        v8f s[4];
#pragma unroll
        for (int tn = 0; tn < 4; ++tn) {
            v16bf b0 = frag_b(Ks, tn * 16 + nn, 72, 0, lane);
            v16bf b1 = frag_b(Ks, tn * 16 + nn, 72, 32, lane);
            v8f z = zero8();
            z = wmma_bf16(a0, b0, z);
            z = wmma_bf16(a1, b1, z);
            s[tn] = z;
        }

        // online softmax per query row (row = r + 8*half, cols across lanes)
#pragma unroll
        for (int r = 0; r < 8; ++r) {
            float mx = -1e30f;
#pragma unroll
            for (int tn = 0; tn < 4; ++tn) {
                s[tn][r] *= 0.125f;            // DH^-0.5
                mx = fmaxf(mx, s[tn][r]);
            }
#pragma unroll
            for (int off = 8; off > 0; off >>= 1)
                mx = fmaxf(mx, __shfl_xor(mx, off, 16));
            float mn = fmaxf(m_run[r], mx);
            float alpha = __expf(m_run[r] - mn);
            float rs = 0.f;
#pragma unroll
            for (int tn = 0; tn < 4; ++tn) {
                float p = __expf(s[tn][r] - mn);
                s[tn][r] = p;
                rs += p;
            }
#pragma unroll
            for (int off = 8; off > 0; off >>= 1)
                rs += __shfl_xor(rs, off, 16);
            l_run[r] = l_run[r] * alpha + rs;
            m_run[r] = mn;
#pragma unroll
            for (int tn = 0; tn < 4; ++tn) o_acc[tn][r] *= alpha;
            int prow = w * 16 + r + 8 * half;  // wave-private P tile in LDS
#pragma unroll
            for (int tn = 0; tn < 4; ++tn)
                Ps[prow * 72 + tn * 16 + nn] = f2bf(s[tn][r]);
        }

        // O += P(16x64) . V ;  B^T = Vt[dh][keys]
        v16bf pa0 = frag_a(Ps, w * 16 + nn, 72, 0, lane);
        v16bf pa1 = frag_a(Ps, w * 16 + nn, 72, 32, lane);
#pragma unroll
        for (int tn = 0; tn < 4; ++tn) {
            v16bf b0 = frag_b(Vts, tn * 16 + nn, 72, 0, lane);
            v16bf b1 = frag_b(Vts, tn * 16 + nn, 72, 32, lane);
            o_acc[tn] = wmma_bf16(pa0, b0, o_acc[tn]);
            o_acc[tn] = wmma_bf16(pa1, b1, o_acc[tn]);
        }
        __syncthreads();
    }

    // finalize: O /= l, scatter heads back into [B, L, D] bf16
#pragma unroll
    for (int r = 0; r < 8; ++r) {
        int q = qblk * 128 + w * 16 + r + 8 * half;
        float inv = 1.0f / l_run[r];
#pragma unroll
        for (int tn = 0; tn < 4; ++tn) {
            float v = o_acc[tn][r] * inv;
            out[((size_t)(b * 2048 + q)) * 1024 + h * 64 + tn * 16 + nn] = f2bf(v);
        }
    }
}

// ---------------------------------------------------------------------------
// Host-side orchestration (graph-capture safe: kernel launches only).
// ---------------------------------------------------------------------------
extern "C" void kernel_launch(void* const* d_in, const int* in_sizes, int n_in,
                              void* d_out, int out_size, void* d_ws, size_t ws_size,
                              hipStream_t stream) {
    (void)in_sizes; (void)n_in; (void)out_size; (void)ws_size;
    const float* x     = (const float*)d_in[0];
    const float* pos   = (const float*)d_in[1];
    const float* ln1_g = (const float*)d_in[2];
    const float* ln1_b = (const float*)d_in[3];
    const float* Wq    = (const float*)d_in[4];
    const float* Wk    = (const float*)d_in[5];
    const float* Wv    = (const float*)d_in[6];
    const float* Wo    = (const float*)d_in[7];
    const float* bo    = (const float*)d_in[8];
    const float* ln2_g = (const float*)d_in[9];
    const float* ln2_b = (const float*)d_in[10];
    const float* W1    = (const float*)d_in[11];
    const float* b1    = (const float*)d_in[12];
    const float* W2    = (const float*)d_in[13];
    const float* b2    = (const float*)d_in[14];

    const size_t MB = 1u << 20;
    char* ws = (char*)d_ws;
    u16*   WqT  = (u16*)(ws + 0 * MB);     //  2 MB  [1024][1024] bf16
    u16*   WkT  = (u16*)(ws + 2 * MB);
    u16*   WvT  = (u16*)(ws + 4 * MB);
    u16*   WoT  = (u16*)(ws + 6 * MB);
    u16*   W1T  = (u16*)(ws + 8 * MB);     //  8 MB  [4096][1024]
    u16*   W2T  = (u16*)(ws + 16 * MB);    //  8 MB  [1024][4096]
    u16*   hbuf = (u16*)(ws + 24 * MB);    //  8 MB  LN1(x)+pos
    u16*   Qbuf = (u16*)(ws + 32 * MB);    //  8 MB  [BH,L,DH]
    u16*   Kbuf = (u16*)(ws + 40 * MB);    //  8 MB
    u16*   Vtb  = (u16*)(ws + 48 * MB);    //  8 MB  [BH,DH,L]
    u16*   AOb  = (u16*)(ws + 56 * MB);    //  8 MB  attention out [M,D]
    float* x2b  = (float*)(ws + 64 * MB);  // 16 MB  residual stream
    u16*   m2b  = (u16*)(ws + 80 * MB);    //  8 MB  LN2 out
    u16*   gbuf = (u16*)(ws + 88 * MB);    // 32 MB  gelu(mW1+b1)  -> 120 MB total

    // --- weight prep: f32 [K][N] -> bf16 [N][K]
    cvt_transpose_kernel<<<(1024 * 1024 + 255) / 256, 256, 0, stream>>>(Wq, WqT, 1024, 1024);
    cvt_transpose_kernel<<<(1024 * 1024 + 255) / 256, 256, 0, stream>>>(Wk, WkT, 1024, 1024);
    cvt_transpose_kernel<<<(1024 * 1024 + 255) / 256, 256, 0, stream>>>(Wv, WvT, 1024, 1024);
    cvt_transpose_kernel<<<(1024 * 1024 + 255) / 256, 256, 0, stream>>>(Wo, WoT, 1024, 1024);
    cvt_transpose_kernel<<<(1024 * 4096 + 255) / 256, 256, 0, stream>>>(W1, W1T, 1024, 4096);
    cvt_transpose_kernel<<<(4096 * 1024 + 255) / 256, 256, 0, stream>>>(W2, W2T, 4096, 1024);

    // --- h = LN1(x)*g + b + pos
    ln_kernel<<<4096, 256, 0, stream>>>(x, ln1_g, ln1_b, pos, hbuf, 1);

    // --- Q/K/V projections (M=4096, N=1024, K=1024)
    dim3 gqkv(1024 / 128, 4096 / 128);
    gemm_bf16_kernel<0><<<gqkv, 256, 0, stream>>>(hbuf, WqT, Qbuf, nullptr, nullptr, 1024, 1024);
    gemm_bf16_kernel<0><<<gqkv, 256, 0, stream>>>(hbuf, WkT, Kbuf, nullptr, nullptr, 1024, 1024);
    gemm_bf16_kernel<1><<<gqkv, 256, 0, stream>>>(hbuf, WvT, Vtb,  nullptr, nullptr, 1024, 1024);

    // --- flash attention: 16 query blocks x 32 (b,h)
    attn_kernel<<<dim3(16, 32), 256, 0, stream>>>(Qbuf, Kbuf, Vtb, AOb);

    // --- x2 = x + attn_out @ Wo + bo
    gemm_bf16_kernel<3><<<gqkv, 256, 0, stream>>>(AOb, WoT, x2b, bo, x, 1024, 1024);

    // --- m = LN2(x2)
    ln_kernel<<<4096, 256, 0, stream>>>(x2b, ln2_g, ln2_b, nullptr, m2b, 0);

    // --- g = gelu(m @ W1 + b1)   (M=4096, N=4096, K=1024)
    gemm_bf16_kernel<2><<<dim3(4096 / 128, 4096 / 128), 256, 0, stream>>>(
        m2b, W1T, gbuf, b1, nullptr, 4096, 1024);

    // --- out = x2 + g @ W2 + b2  (M=4096, N=1024, K=4096)
    gemm_bf16_kernel<3><<<dim3(1024 / 128, 4096 / 128), 256, 0, stream>>>(
        gbuf, W2T, (float*)d_out, b2, x2b, 1024, 4096);
}
